// ClusterLoss_6511170421412
// MI455X (gfx1250) — matrix operand order repaired and checked by reading, not compiled, not run on hardware
//
#include <hip/hip_runtime.h>
#include <hip/hip_bf16.h>

// Problem constants (match reference setup_inputs)
#define PB 512      // batch
#define PK 64       // frames per batch
#define PD 512      // feature dim
#define PN 1024     // num centers
#define PM (PB*PK)  // total rows = 32768
#define NT (PN/16)  // 64 center tiles

typedef __attribute__((ext_vector_type(16))) __bf16 v16bf;
typedef __attribute__((ext_vector_type(8)))  float  v8f;

union AFrag { v16bf v; unsigned short u[16]; };

static __device__ __forceinline__ unsigned short f2bf(float f) {
    unsigned int x = __float_as_uint(f);
    x += 0x7FFFu + ((x >> 16) & 1u);       // round-to-nearest-even
    return (unsigned short)(x >> 16);
}

// ---------------------------------------------------------------------------
// Pass 0: zero coverage bitmask, precompute 0.5*||c_n||^2, convert centers
//         to a bf16 image in workspace (so the GEMM pass can DMA it to LDS).
// ---------------------------------------------------------------------------
__global__ void cl_init_kernel(const float* __restrict__ centers,
                               float* __restrict__ halfCsq,
                               unsigned* __restrict__ bits,
                               unsigned short* __restrict__ cbf) {
    const int t = blockIdx.x * blockDim.x + threadIdx.x;   // 16384 threads
    if (t < PB * (PN / 32)) bits[t] = 0u;
    if (t < PN) {
        const float* c = centers + (size_t)t * PD;
        float s = 0.f;
        for (int d = 0; d < PD; ++d) s += c[d] * c[d];
        halfCsq[t] = 0.5f * s;
    }
    for (int i = t; i < PN * PD; i += 16384)
        cbf[i] = f2bf(centers[i]);
}

// ---------------------------------------------------------------------------
// Pass 1: per-row argmax(x.c - 0.5*c^2) via bf16 WMMA, set coverage bits.
// Block = 256 threads = 8 waves; wave w owns rows [blk*128 + w*16, +16).
// A panel lives in registers (X read once from HBM); B tiles are bf16 and
// double-buffered in LDS via CDNA5 async global->LDS copies (ASYNCcnt).
// ---------------------------------------------------------------------------
__global__ void __launch_bounds__(256, 1)
cl_assign_kernel(const float* __restrict__ X,
                 const unsigned short* __restrict__ cbf,   // bf16 centers [N,D]
                 const float* __restrict__ halfCsq,
                 unsigned* __restrict__ bits) {
    // double-buffered B tile: 2 x (16 centers x 512 d) bf16 = 2 x 16 KB
    __shared__ unsigned short bSh[2][16 * PD];

    const int lane = threadIdx.x & 31;
    const int wave = threadIdx.x >> 5;
    const int half = (lane >> 4) & 1;                 // wave32 lane half
    const int mrow = lane & 15;
    const int m0   = blockIdx.x * 128 + wave * 16;    // this wave's M tile

    // ---- Preload A panel: 16 k-fragments (16x32 bf16 each) in registers ----
    // A layout (matches B): lanes 0-15 hold K = k0+0..15,
    //                      lanes 16-31 hold K = k0+16..31, row = lane%16
    AFrag aF[16];
    const float* xRow = X + (size_t)(m0 + mrow) * PD;
    #pragma unroll
    for (int ks = 0; ks < 16; ++ks) {
        const int base = ks * 32 + half * 16;
        #pragma unroll
        for (int j = 0; j < 8; ++j) aF[ks].u[j]     = f2bf(xRow[base + j]);
        #pragma unroll
        for (int j = 0; j < 8; ++j) aF[ks].u[8 + j] = f2bf(xRow[base + 8 + j]);
    }

    // ---- async-stage helper: copy tile nt (16x512 bf16 = 16 KB, contiguous
    //      in cbf) into LDS buffer `buf`; 4 x b128 per thread.
    // LDS destination operand is derived from the real shared pointer
    // (ptrtoint of generic LDS address == wave-relative LDS byte address on
    // gfx1250); this also forces bSh to escape so the "memory" clobber makes
    // the compiler reload the B fragments after the async copies land.
    auto stage_async = [&](int nt, int buf) {
        const char* gsrc = (const char*)(cbf + (size_t)nt * (16 * PD));
        const unsigned ldsBase =
            (unsigned)(uintptr_t)(const void*)&bSh[buf][0];
        #pragma unroll
        for (int i = 0; i < 4; ++i) {
            const unsigned off = threadIdx.x * 16u + (unsigned)i * 4096u;
            const unsigned lds = ldsBase + off;
            const void* g = gsrc + off;
            asm volatile("global_load_async_to_lds_b128 %0, %1, off"
                         :: "v"(lds), "v"(g) : "memory");
        }
    };

    float bestV[8];
    int   bestI[8];
    #pragma unroll
    for (int r = 0; r < 8; ++r) { bestV[r] = -3.0e38f; bestI[r] = 0; }

    stage_async(0, 0);                                 // prologue: tile 0

    for (int nt = 0; nt < NT; ++nt) {
        const int cur = nt & 1;
        if (nt + 1 < NT) {
            stage_async(nt + 1, 1 - cur);              // prefetch next tile
            asm volatile("s_wait_asynccnt 0x4" ::: "memory");  // tile nt landed
        } else {
            asm volatile("s_wait_asynccnt 0x0" ::: "memory");
        }
        __syncthreads();                               // tile nt visible to all

        // two interleaved accumulators break the WMMA RAW chain (ILP x2)
        v8f acc0 = {}, acc1 = {};
        const unsigned short* bRow = &bSh[cur][(lane & 15) * PD];
        #pragma unroll
        for (int kp = 0; kp < 8; ++kp) {
            // B layout (same k order as A): lane = N column (mod 16); lanes 0-15 hold
            // K = k0+0..15, lanes 16-31 hold K = k0+16..31 (contiguous run).
            const v16bf b0 = *(const v16bf*)&bRow[(2 * kp + 0) * 32 + half * 16];
            const v16bf b1 = *(const v16bf*)&bRow[(2 * kp + 1) * 32 + half * 16];
            acc0 = __builtin_amdgcn_wmma_f32_16x16x32_bf16(
                false, aF[2 * kp + 0].v, false, b0, (short)0, acc0, false, false);
            acc1 = __builtin_amdgcn_wmma_f32_16x16x32_bf16(
                false, aF[2 * kp + 1].v, false, b1, (short)0, acc1, false, false);
        }

        // score = cross - 0.5*||c||^2 ; argmin dist == argmax score
        const int   n0   = nt * 16;
        const int   nIdx = n0 + (lane & 15);
        const float hcs  = halfCsq[nIdx];
        #pragma unroll
        for (int r = 0; r < 8; ++r) {
            const float s = (acc0[r] + acc1[r]) - hcs;
            if (s > bestV[r]) { bestV[r] = s; bestI[r] = nIdx; }
        }
        __syncthreads();            // reads done before next overwrite of buf
    }

    // ---- reduce argmax across the 16 lanes sharing each M row ----
    #pragma unroll
    for (int msk = 1; msk < 16; msk <<= 1) {
        #pragma unroll
        for (int r = 0; r < 8; ++r) {
            const float ov = __shfl_xor(bestV[r], msk, 32);
            const int   oi = __shfl_xor(bestI[r], msk, 32);
            if (ov > bestV[r] || (ov == bestV[r] && oi < bestI[r])) {
                bestV[r] = ov; bestI[r] = oi;
            }
        }
    }

    // lanes 0 and 16 own rows m0+r and m0+8+r respectively (C/D layout)
    if (mrow == 0) {
        #pragma unroll
        for (int r = 0; r < 8; ++r) {
            const int m   = m0 + half * 8 + r;
            const int b   = m >> 6;                    // / K(=64)
            const int idx = bestI[r];
            atomicOr(&bits[b * (PN / 32) + (idx >> 5)], 1u << (idx & 31));
        }
    }
}

// ---------------------------------------------------------------------------
// Pass 2: coverage popcount over batches + entropy reduction (1 block).
// ---------------------------------------------------------------------------
__global__ void cl_entropy_kernel(const unsigned* __restrict__ bits,
                                  float* __restrict__ out) {
    __shared__ float red[1024];
    const int n = threadIdx.x;                         // blockDim.x == 1024 == PN
    const int word = n >> 5;
    const int bit  = n & 31;
    int cov = 0;
    for (int b = 0; b < PB; ++b)
        cov += (int)((bits[b * (PN / 32) + word] >> bit) & 1u);
    const float p = (float)cov * (1.0f / (float)(PB * PK));
    red[n] = -p * logf(p + 1e-10f);
    __syncthreads();
    for (int s = 512; s > 0; s >>= 1) {
        if (n < s) red[n] += red[n + s];
        __syncthreads();
    }
    if (n == 0) out[0] = red[0];
}

// ---------------------------------------------------------------------------
extern "C" void kernel_launch(void* const* d_in, const int* in_sizes, int n_in,
                              void* d_out, int out_size, void* d_ws, size_t ws_size,
                              hipStream_t stream) {
    const float* X  = (const float*)d_in[0];   // [B,K,D] -> [M,D]
    const float* Cc = (const float*)d_in[1];   // [N,D]
    float* out = (float*)d_out;

    float*          halfCsq = (float*)d_ws;                          // 4 KB
    unsigned*       bits    = (unsigned*)((char*)d_ws + 4096);       // 64 KB
    unsigned short* cbf     = (unsigned short*)((char*)d_ws + 69632);// 1 MB bf16 centers

    cl_init_kernel<<<64, 256, 0, stream>>>(Cc, halfCsq, bits, cbf);
    cl_assign_kernel<<<PM / 128, 256, 0, stream>>>(X, cbf, halfCsq, bits);
    cl_entropy_kernel<<<1, 1024, 0, stream>>>(bits, out);
}
